// MPNNLayer_70334384439335
// MI455X (gfx1250) — compile-verified, matrix operand order checked by reading, not run on hardware
//
#include <hip/hip_runtime.h>

// ---------------------------------------------------------------------------
// Types / helpers
// ---------------------------------------------------------------------------
typedef __attribute__((ext_vector_type(16))) __bf16 v16bf;
typedef __attribute__((ext_vector_type(2)))  __bf16 v2bf;
typedef __attribute__((ext_vector_type(8)))  float  v8f;
typedef __attribute__((ext_vector_type(4)))  float  v4f;

#define WMMA_BF16(a, b, c) \
  __builtin_amdgcn_wmma_f32_16x16x32_bf16(false, (a), false, (b), (short)0, (c), false, false)

// native hardware converts (RNE) instead of manual integer rounding
__device__ __forceinline__ unsigned short f2bf(float f) {
  __bf16 b = (__bf16)f;
  return __builtin_bit_cast(unsigned short, b);
}
__device__ __forceinline__ unsigned int pack2bf(float f0, float f1) {
  v2bf v;
  v.x = (__bf16)f0;
  v.y = (__bf16)f1;
  return __builtin_bit_cast(unsigned int, v);
}

// A/B bf16 fragment slot -> k_local (0..31) mapping.
// lane<16 holds k in {0..7,16..23}, lane>=16 holds k in {8..15,24..31}.
__device__ __forceinline__ int slot_to_klocal(int lane, int j) {
  int v = j >> 1, p = j & 1;
  int bk = (v < 4) ? (2 * v) : (16 + 2 * (v - 4));
  return bk + p + ((lane < 16) ? 0 : 8);
}
// Inverse: element (m in 0..15, klocal in 0..31) -> (lane, j) in fragment.
__device__ __forceinline__ void elem_to_slot(int m, int klocal, int& lane, int& j) {
  int hi = (klocal >> 3) & 1;
  int kk = klocal & ~8;                 // in {0..7} U {16..23}
  lane = m + (hi ? 16 : 0);
  j = (kk < 8) ? kk : (kk - 8);
}

// Packed-weight region offsets (bf16 elements) inside workspace
#define W1P_OFF 0          // 384x128 -> 12x8 tiles
#define W2P_OFF 49152      // 128x128 -> 4x8
#define W3P_OFF 65536      // 128x128 -> 4x8
#define F1P_OFF 81920      // 128x512 -> 4x32
#define F2P_OFF 147456     // 512x128 -> 16x8
#define PACK_TOTAL 212992

#define N_NODES 30000
#define N_EDGES 480000

// ---------------------------------------------------------------------------
// Kernel 0: pack weights into bf16 WMMA-B fragment tiles
// ---------------------------------------------------------------------------
__global__ void pack_weights_kernel(const float* __restrict__ W1,
                                    const float* __restrict__ W2,
                                    const float* __restrict__ W3,
                                    const float* __restrict__ F1,
                                    const float* __restrict__ F2,
                                    unsigned short* __restrict__ P) {
  int t = blockIdx.x * 256 + threadIdx.x;
  if (t >= PACK_TOTAL) return;
  const float* src;
  int Nn, off;
  if      (t < W2P_OFF) { src = W1; Nn = 128; off = W1P_OFF; }
  else if (t < W3P_OFF) { src = W2; Nn = 128; off = W2P_OFF; }
  else if (t < F1P_OFF) { src = W3; Nn = 128; off = W3P_OFF; }
  else if (t < F2P_OFF) { src = F1; Nn = 512; off = F1P_OFF; }
  else                  { src = F2; Nn = 128; off = F2P_OFF; }
  int s    = t - off;
  int tile = s >> 9;
  int r    = s & 511;
  int lane = r >> 4;
  int j    = r & 15;
  int ntn  = Nn >> 4;
  int kt   = tile / ntn;
  int nt   = tile % ntn;
  int k = kt * 32 + slot_to_klocal(lane, j);
  int n = nt * 16 + (lane & 15);
  P[t] = f2bf(src[(size_t)k * Nn + n]);
}

// ---------------------------------------------------------------------------
// Kernel 1: in-degree
// ---------------------------------------------------------------------------
__global__ void degree_kernel(const int* __restrict__ tgt, float* __restrict__ deg, int E) {
  int e = blockIdx.x * 256 + threadIdx.x;
  if (e < E) atomicAdd(&deg[tgt[e]], 1.0f);
}

// ---------------------------------------------------------------------------
// Kernel 2: edge MLP (3 layers) + scatter-add into dh.
// 128 edges per block, 8 waves, wave w owns rows [w*16, w*16+16).
// ---------------------------------------------------------------------------
__global__ void __launch_bounds__(256)
edge_mlp_kernel(const float* __restrict__ hV, const float* __restrict__ hE,
                const int* __restrict__ src, const int* __restrict__ tgt,
                const float* __restrict__ b1, const float* __restrict__ b2,
                const float* __restrict__ b3,
                const unsigned short* __restrict__ Pw,
                float* __restrict__ dh) {
  __shared__ __align__(64) unsigned short astage[128 * 32];   // 8 KB: one K-slab of A
  __shared__ __align__(64) unsigned short mbuf[128 * 128];    // 32 KB: activations, A-layout

  const int tid = threadIdx.x;
  const int w   = tid >> 5;
  const int l   = tid & 31;
  const int e0  = blockIdx.x * 128;

  // staging: this thread always fills fragment slots [tid*16, tid*16+16)
  const int sbase  = tid * 16;
  const int s_rt   = sbase >> 9;
  const int s_lane = (sbase & 511) >> 4;
  const int s_row  = s_rt * 16 + (s_lane & 15);
  const int s_k8   = (s_lane < 16) ? 0 : 8;
  const int s_e    = e0 + s_row;
  const float* rowE = hE + (size_t)s_e * 128;
  const float* rowS = hV + (size_t)src[s_e] * 128;
  const float* rowT = hV + (size_t)tgt[s_e] * 128;

  const unsigned short* W1p = Pw + W1P_OFF;
  const unsigned short* W2p = Pw + W2P_OFF;
  const unsigned short* W3p = Pw + W3P_OFF;

  v8f acc[8] = {};

  // ---- layer 1: [128,384] x [384,128] ----
  for (int ks = 0; ks < 12; ++ks) {
    const int k0 = ks * 32;
    const float* rp = (k0 < 128) ? (rowE + k0)
                    : (k0 < 256) ? (rowS + (k0 - 128))
                                 : (rowT + (k0 - 256));
#pragma unroll
    for (int v = 0; v < 8; ++v) {
      int bk = (v < 4) ? (2 * v) : (16 + 2 * (v - 4));
      int kl = bk + s_k8;                       // pair (kl, kl+1), stays in-slab
      *(unsigned int*)&astage[sbase + 2 * v] = pack2bf(rp[kl], rp[kl + 1]);
    }
    __syncthreads();
    v16bf a = *(const v16bf*)&astage[w * 512 + l * 16];
#pragma unroll
    for (int n = 0; n < 8; ++n) {
      v16bf b = *(const v16bf*)&W1p[(ks * 8 + n) * 512 + l * 16];
      acc[n] = WMMA_BF16(a, b, acc[n]);
    }
    __syncthreads();
  }

  // bias + relu -> mbuf (A-fragment layout; wave-private tiles (w*4+kt))
  const int colb = l & 15;
  const int mb   = (l < 16) ? 0 : 8;
#pragma unroll
  for (int n = 0; n < 8; ++n) {
    int c  = n * 16 + colb;
    float bias = b1[c];
    int kt = c >> 5, klocal = c & 31;
#pragma unroll
    for (int i = 0; i < 8; ++i) {
      float val = acc[n][i] + bias;
      val = val > 0.f ? val : 0.f;
      int lanep, j;
      elem_to_slot(mb + i, klocal, lanep, j);
      mbuf[(w * 4 + kt) * 512 + lanep * 16 + j] = f2bf(val);
    }
    acc[n] = (v8f){};
  }

  // ---- layer 2 ----
#pragma unroll
  for (int kt = 0; kt < 4; ++kt) {
    v16bf a = *(const v16bf*)&mbuf[(w * 4 + kt) * 512 + l * 16];
#pragma unroll
    for (int n = 0; n < 8; ++n) {
      v16bf b = *(const v16bf*)&W2p[(kt * 8 + n) * 512 + l * 16];
      acc[n] = WMMA_BF16(a, b, acc[n]);
    }
  }
#pragma unroll
  for (int n = 0; n < 8; ++n) {
    int c  = n * 16 + colb;
    float bias = b2[c];
    int kt = c >> 5, klocal = c & 31;
#pragma unroll
    for (int i = 0; i < 8; ++i) {
      float val = acc[n][i] + bias;
      val = val > 0.f ? val : 0.f;
      int lanep, j;
      elem_to_slot(mb + i, klocal, lanep, j);
      mbuf[(w * 4 + kt) * 512 + lanep * 16 + j] = f2bf(val);
    }
    acc[n] = (v8f){};
  }

  // ---- layer 3 (no relu) ----
#pragma unroll
  for (int kt = 0; kt < 4; ++kt) {
    v16bf a = *(const v16bf*)&mbuf[(w * 4 + kt) * 512 + l * 16];
#pragma unroll
    for (int n = 0; n < 8; ++n) {
      v16bf b = *(const v16bf*)&W3p[(kt * 8 + n) * 512 + l * 16];
      acc[n] = WMMA_BF16(a, b, acc[n]);
    }
  }

  // ---- scatter-add into dh (L2-resident float atomics) ----
  int trow[8];
#pragma unroll
  for (int i = 0; i < 8; ++i) trow[i] = tgt[e0 + w * 16 + mb + i];
#pragma unroll
  for (int n = 0; n < 8; ++n) {
    int c = n * 16 + colb;
    float bias = b3[c];
#pragma unroll
    for (int i = 0; i < 8; ++i) {
      atomicAdd(&dh[(size_t)trow[i] * 128 + c], acc[n][i] + bias);
    }
  }
}

// ---------------------------------------------------------------------------
// Kernel 3: node update: mean-aggregate + LN0 + FFN(128->512->128) + LN1.
// 128 nodes per block, 8 waves.
// ---------------------------------------------------------------------------
#define XPITCH 132   // keeps float4 LDS accesses 16B-aligned
__global__ void __launch_bounds__(256)
node_kernel(const float* __restrict__ hV, const float* __restrict__ dh,
            const float* __restrict__ deg,
            const float* __restrict__ ln0w, const float* __restrict__ ln0b,
            const float* __restrict__ ln1w, const float* __restrict__ ln1b,
            const float* __restrict__ fb1, const float* __restrict__ fb2,
            const unsigned short* __restrict__ Pw,
            float* __restrict__ out, int N) {
  __shared__ __align__(64) float          xbuf[128 * XPITCH];  // 66 KB f32
  __shared__ __align__(64) unsigned short abuf[128 * 128];     // 32 KB: h bf16, A-layout
  __shared__ __align__(64) unsigned short tbuf[128 * 128];     // 32 KB: FFN chunk, A-layout

  const int tid = threadIdx.x;
  const int w   = tid >> 5;
  const int l   = tid & 31;
  const int n0  = blockIdx.x * 128;

  const unsigned short* F1p = Pw + F1P_OFF;
  const unsigned short* F2p = Pw + F2P_OFF;

  // stage 1: x = h_V + dh/deg   (float4 path: 4096 vec4 elements, 16/thread)
#pragma unroll 4
  for (int i = 0; i < 16; ++i) {
    int flat = i * 256 + tid;         // vec4 index
    int row = flat >> 5, c4 = flat & 31;
    int g = n0 + row;
    g = (g < N) ? g : (N - 1);
    float inv = 1.0f / fmaxf(deg[g], 1.0f);
    v4f hv = *(const v4f*)(hV + (size_t)g * 128 + c4 * 4);
    v4f dv = *(const v4f*)(dh + (size_t)g * 128 + c4 * 4);
    v4f x  = hv + dv * inv;
    *(v4f*)&xbuf[row * XPITCH + c4 * 4] = x;
  }
  __syncthreads();

  // stage 2: LN0 per row; write h back to xbuf + packed bf16 pairs to abuf
  if (tid < 128) {
    const int row = tid;
    float m = 0.f;
    for (int c = 0; c < 128; ++c) m += xbuf[row * XPITCH + c];
    m *= (1.0f / 128.0f);
    float v = 0.f;
    for (int c = 0; c < 128; ++c) { float d = xbuf[row * XPITCH + c] - m; v += d * d; }
    v *= (1.0f / 128.0f);
    float rs = rsqrtf(v + 1e-5f);
    const int rt = row >> 4, mloc = row & 15;
    for (int c = 0; c < 128; c += 2) {
      float h0 = (xbuf[row * XPITCH + c]     - m) * rs * ln0w[c]     + ln0b[c];
      float h1 = (xbuf[row * XPITCH + c + 1] - m) * rs * ln0w[c + 1] + ln0b[c + 1];
      xbuf[row * XPITCH + c]     = h0;
      xbuf[row * XPITCH + c + 1] = h1;
      // even c: (c, c+1) occupy consecutive slots of the same lane
      int lanep, j;
      elem_to_slot(mloc, c & 31, lanep, j);
      *(unsigned int*)&abuf[(rt * 4 + (c >> 5)) * 512 + lanep * 16 + j] = pack2bf(h0, h1);
    }
  }
  __syncthreads();

  // stage 3: FFN via WMMA, 4 chunks of 128 intermediate columns
  const int colb = l & 15;
  const int mb   = (l < 16) ? 0 : 8;
  v8f oacc[8] = {};
  for (int ch = 0; ch < 4; ++ch) {
    v8f tacc[8] = {};
#pragma unroll
    for (int kt = 0; kt < 4; ++kt) {
      v16bf a = *(const v16bf*)&abuf[(w * 4 + kt) * 512 + l * 16];
#pragma unroll
      for (int n = 0; n < 8; ++n) {
        v16bf b = *(const v16bf*)&F1p[(kt * 32 + ch * 8 + n) * 512 + l * 16];
        tacc[n] = WMMA_BF16(a, b, tacc[n]);
      }
    }
#pragma unroll
    for (int n = 0; n < 8; ++n) {
      int c = n * 16 + colb;
      float bias = fb1[ch * 128 + c];
      int kt = c >> 5, klocal = c & 31;
#pragma unroll
      for (int i = 0; i < 8; ++i) {
        float val = tacc[n][i] + bias;
        val = val > 0.f ? val : 0.f;
        int lanep, j;
        elem_to_slot(mb + i, klocal, lanep, j);
        tbuf[(w * 4 + kt) * 512 + lanep * 16 + j] = f2bf(val);   // wave-private tiles
      }
    }
#pragma unroll
    for (int kt = 0; kt < 4; ++kt) {
      v16bf a = *(const v16bf*)&tbuf[(w * 4 + kt) * 512 + l * 16];
#pragma unroll
      for (int n = 0; n < 8; ++n) {
        v16bf b = *(const v16bf*)&F2p[((ch * 4 + kt) * 8 + n) * 512 + l * 16];
        oacc[n] = WMMA_BF16(a, b, oacc[n]);
      }
    }
  }

  // residual: y = h + dh2 + b2, written back into xbuf
#pragma unroll
  for (int n = 0; n < 8; ++n) {
    int c = n * 16 + colb;
    float bias = fb2[c];
#pragma unroll
    for (int i = 0; i < 8; ++i) {
      int row = w * 16 + mb + i;
      xbuf[row * XPITCH + c] += oacc[n][i] + bias;
    }
  }
  __syncthreads();

  // stage 4: LN1 per row -> global out (float4 stores)
  if (tid < 128) {
    const int row = tid;
    const int g = n0 + row;
    float m = 0.f;
    for (int c = 0; c < 128; ++c) m += xbuf[row * XPITCH + c];
    m *= (1.0f / 128.0f);
    float v = 0.f;
    for (int c = 0; c < 128; ++c) { float d = xbuf[row * XPITCH + c] - m; v += d * d; }
    v *= (1.0f / 128.0f);
    float rs = rsqrtf(v + 1e-5f);
    if (g < N) {
      for (int c4 = 0; c4 < 32; ++c4) {
        v4f x = *(const v4f*)&xbuf[row * XPITCH + c4 * 4];
        v4f gw = *(const v4f*)(ln1w + c4 * 4);
        v4f gb = *(const v4f*)(ln1b + c4 * 4);
        v4f y = (x - m) * rs * gw + gb;
        *(v4f*)(out + (size_t)g * 128 + c4 * 4) = y;
      }
    }
  }
}

// ---------------------------------------------------------------------------
// Launch
// ---------------------------------------------------------------------------
extern "C" void kernel_launch(void* const* d_in, const int* in_sizes, int n_in,
                              void* d_out, int out_size, void* d_ws, size_t ws_size,
                              hipStream_t stream) {
  const int N = N_NODES, E = N_EDGES;
  const float* h_V   = (const float*)d_in[0];
  const float* h_E   = (const float*)d_in[1];
  const int*   src   = (const int*)d_in[2];
  const int*   tgt   = (const int*)d_in[3];
  const float* W1    = (const float*)d_in[4];
  const float* b1    = (const float*)d_in[5];
  const float* W2    = (const float*)d_in[6];
  const float* b2    = (const float*)d_in[7];
  const float* W3    = (const float*)d_in[8];
  const float* b3    = (const float*)d_in[9];
  const float* ln0w  = (const float*)d_in[10];
  const float* ln0b  = (const float*)d_in[11];
  const float* ln1w  = (const float*)d_in[12];
  const float* ln1b  = (const float*)d_in[13];
  const float* F1    = (const float*)d_in[14];
  const float* fb1   = (const float*)d_in[15];
  const float* F2    = (const float*)d_in[16];
  const float* fb2   = (const float*)d_in[17];
  float* out = (float*)d_out;

  float* dh  = (float*)d_ws;                       // N*128 f32
  float* deg = dh + (size_t)N * 128;               // N f32
  unsigned short* Pw = (unsigned short*)(deg + N); // PACK_TOTAL bf16

  hipMemsetAsync(d_ws, 0, ((size_t)N * 128 + N) * sizeof(float), stream);

  pack_weights_kernel<<<(PACK_TOTAL + 255) / 256, 256, 0, stream>>>(W1, W2, W3, F1, F2, Pw);
  degree_kernel<<<(E + 255) / 256, 256, 0, stream>>>(tgt, deg, E);
  edge_mlp_kernel<<<E / 128, 256, 0, stream>>>(h_V, h_E, src, tgt, b1, b2, b3, Pw, dh);
  node_kernel<<<(N + 127) / 128, 256, 0, stream>>>(h_V, dh, deg, ln0w, ln0b, ln1w, ln1b,
                                                   fb1, fb2, Pw, out, N);
}
